// VQ_27169963114912
// MI455X (gfx1250) — matrix operand order repaired and checked by reading, not compiled, not run on hardware
//
#include <hip/hip_runtime.h>

// ---------------------------------------------------------------------------
// VQ (vector-quantization) argmin + gather + loss for MI455X (gfx1250).
//   z:        (32768, 256) fp32
//   codebook: ( 8192, 256) fp32
//   out:      z_q (32768*256 fp32) ++ loss (1 fp32)
//
// argmin_k ||z - e_k||^2 == argmin_k ( ||e_k||^2 - 2 z.e_k )   (||z||^2 const)
// Matmul z @ E^T done with bf16 hi/lo split (3x v_wmma_f32_16x16x32_bf16 per
// 32-wide K chunk, fp32 accumulate) for fp32-grade argmin fidelity.
// ---------------------------------------------------------------------------

#define VQ_DIM   256
#define VQ_K     8192
#define VQ_N     32768

typedef __attribute__((ext_vector_type(16))) __bf16 v16bf;
typedef __attribute__((ext_vector_type(8)))  float  v8f;

union B16x16 { uint4 q[2]; v16bf v; };

static __device__ inline v16bf load_frag(const unsigned short* p0,
                                         const unsigned short* p1) {
  B16x16 u;
  u.q[0] = *reinterpret_cast<const uint4*>(p0);
  u.q[1] = *reinterpret_cast<const uint4*>(p1);
  return u.v;
}

// --- fp32 -> (bf16 hi, bf16 lo) split, RNE rounding --------------------------
__global__ void vq_split_kernel(const float* __restrict__ src,
                                unsigned short* __restrict__ hi,
                                unsigned short* __restrict__ lo, int n) {
  int i = blockIdx.x * blockDim.x + threadIdx.x;
  int stride = gridDim.x * blockDim.x;
  for (; i < n; i += stride) {
    float f = src[i];
    unsigned u = __float_as_uint(f);
    unsigned r = u + 0x7FFFu + ((u >> 16) & 1u);
    unsigned short h = (unsigned short)(r >> 16);
    float hf = __uint_as_float(((unsigned)h) << 16);
    float l = f - hf;
    unsigned ul = __float_as_uint(l);
    unsigned rl = ul + 0x7FFFu + ((ul >> 16) & 1u);
    hi[i] = h;
    lo[i] = (unsigned short)(rl >> 16);
  }
}

// --- ||e_k||^2 in fp32 (exact, from the original codebook) -------------------
__global__ __launch_bounds__(256) void vq_esq_kernel(
    const float* __restrict__ cb, float* __restrict__ e_sq) {
  __shared__ float s[256];
  const int row = blockIdx.x;
  float v = cb[(size_t)row * VQ_DIM + threadIdx.x];
  s[threadIdx.x] = v * v;
  __syncthreads();
  for (int st = 128; st > 0; st >>= 1) {
    if ((int)threadIdx.x < st) s[threadIdx.x] += s[threadIdx.x + st];
    __syncthreads();
  }
  if (threadIdx.x == 0) e_sq[row] = s[0];
}

// --- main WMMA argmin kernel -------------------------------------------------
// 256 threads = 8 waves; each wave owns a 16-row tile of z and sweeps all
// 8192 codes in 16-column B tiles, K=256 in 8 chunks of 32 (bf16 WMMA).
__global__ __launch_bounds__(256) void vq_argmin_kernel(
    const unsigned short* __restrict__ z_hi,
    const unsigned short* __restrict__ z_lo,
    const unsigned short* __restrict__ cb_hi,
    const unsigned short* __restrict__ cb_lo,
    const float* __restrict__ e_sq,
    int* __restrict__ out_idx) {
  const int lane = threadIdx.x & 31;
  const int wave = threadIdx.x >> 5;
  const int rowBase = blockIdx.x * 128 + wave * 16;
  const int half = lane >> 4;   // 0: lanes 0-15, 1: lanes 16-31
  const int l15  = lane & 15;

  // A-matrix 16x32 bf16 layout (ISA 7.12.2): lanes 0-15 row M=lane, K runs
  // [0..7],[16..23]; lanes 16-31 row M=lane-16, K runs [8..15],[24..31].
  const size_t arow = (size_t)(rowBase + l15) * VQ_DIM;
  v16bf a_hi[8], a_lo[8];
#pragma unroll
  for (int c = 0; c < 8; ++c) {
    const int kb = c * 32 + (half ? 8 : 0);
    const unsigned short* p0 = z_hi + arow + kb;
    const unsigned short* q0 = z_lo + arow + kb;
    a_hi[c] = load_frag(p0, p0 + 16);
    a_lo[c] = load_frag(q0, q0 + 16);
  }

  float bestv[8];
  int   besti[8];
#pragma unroll
  for (int r = 0; r < 8; ++r) { bestv[r] = 3.0e38f; besti[r] = 0; }

  for (int colBase = 0; colBase < VQ_K; colBase += 16) {
    const int col = colBase + l15;
    // B-matrix 32x16 bf16 layout: lane holds column N=l15; lanes 0-15 carry
    // K runs [0..7],[16..23]; lanes 16-31 carry K runs [8..15],[24..31].
    const unsigned short* brow_hi = cb_hi + (size_t)col * VQ_DIM + half * 8;
    const unsigned short* brow_lo = cb_lo + (size_t)col * VQ_DIM + half * 8;
    // Prefetch next column tile (emits global_prefetch_b8).
    __builtin_prefetch(brow_hi + (size_t)16 * VQ_DIM, 0, 1);

    v8f acc = {};
#pragma unroll
    for (int c = 0; c < 8; ++c) {
      const int kb = c * 32;
      v16bf b_hi = load_frag(brow_hi + kb, brow_hi + kb + 16);
      v16bf b_lo = load_frag(brow_lo + kb, brow_lo + kb + 16);
      acc = __builtin_amdgcn_wmma_f32_16x16x32_bf16(false, a_hi[c], false, b_hi,
                                                    (short)0, acc, false, false);
      acc = __builtin_amdgcn_wmma_f32_16x16x32_bf16(false, a_hi[c], false, b_lo,
                                                    (short)0, acc, false, false);
      acc = __builtin_amdgcn_wmma_f32_16x16x32_bf16(false, a_lo[c], false, b_hi,
                                                    (short)0, acc, false, false);
    }

    // C/D layout: VGPR r, lanes 0-15 -> row rowBase+r, lanes 16-31 -> +8.
    const float es = e_sq[col];
#pragma unroll
    for (int r = 0; r < 8; ++r) {
      float d = es - 2.0f * acc[r];
      if (d < bestv[r]) { bestv[r] = d; besti[r] = col; }
    }
  }

  // Cross-lane argmin within each 16-lane half (xor masks 8,4,2,1 stay inside
  // the half). Tie-break on lower index to match jnp.argmin.
#pragma unroll
  for (int r = 0; r < 8; ++r) {
    float bv = bestv[r];
    int   bi = besti[r];
#pragma unroll
    for (int off = 8; off >= 1; off >>= 1) {
      float ov = __shfl_xor(bv, off, 32);
      int   oi = __shfl_xor(bi, off, 32);
      if (ov < bv || (ov == bv && oi < bi)) { bv = ov; bi = oi; }
    }
    if (l15 == 0) out_idx[rowBase + half * 8 + r] = bi;
  }
}

// --- gather z_q + per-block partial of sum((z_q - z)^2) ----------------------
__global__ __launch_bounds__(256) void vq_gather_loss_kernel(
    const float* __restrict__ z, const float* __restrict__ cb,
    const int* __restrict__ idx, float* __restrict__ out,
    float* __restrict__ partials) {
  __shared__ float s[256];
  const size_t base = (size_t)blockIdx.x * 8192 + (size_t)threadIdx.x * 32;
  const int n = (int)(base >> 8);      // row (32 | 256, so one row per thread)
  const float* crow = cb + (size_t)idx[n] * VQ_DIM + (base & 255);
  const float* zrow = z + base;
  float* orow = out + base;
  float acc = 0.f;
#pragma unroll
  for (int j = 0; j < 8; ++j) {
    float4 c4 = reinterpret_cast<const float4*>(crow)[j];
    float4 z4 = reinterpret_cast<const float4*>(zrow)[j];
    float dx = c4.x - z4.x, dy = c4.y - z4.y, dz = c4.z - z4.z, dw = c4.w - z4.w;
    acc += dx * dx + dy * dy + dz * dz + dw * dw;
    reinterpret_cast<float4*>(orow)[j] = c4;
  }
  s[threadIdx.x] = acc;
  __syncthreads();
  for (int st = 128; st > 0; st >>= 1) {
    if ((int)threadIdx.x < st) s[threadIdx.x] += s[threadIdx.x + st];
    __syncthreads();
  }
  if (threadIdx.x == 0) partials[blockIdx.x] = s[0];
}

// --- deterministic final reduction: loss = (1+beta) * sum / (N*D) ------------
__global__ __launch_bounds__(1024) void vq_finalize_kernel(
    const float* __restrict__ partials, float* __restrict__ out_loss) {
  __shared__ float s[1024];
  s[threadIdx.x] = partials[threadIdx.x];
  __syncthreads();
  for (int st = 512; st > 0; st >>= 1) {
    if ((int)threadIdx.x < st) s[threadIdx.x] += s[threadIdx.x + st];
    __syncthreads();
  }
  if (threadIdx.x == 0)
    out_loss[0] = s[0] * (1.25f / 8388608.0f);  // (1+0.25) * mean over N*D
}

extern "C" void kernel_launch(void* const* d_in, const int* in_sizes, int n_in,
                              void* d_out, int out_size, void* d_ws, size_t ws_size,
                              hipStream_t stream) {
  const float* z  = (const float*)d_in[0];   // 32768*256
  const float* cb = (const float*)d_in[1];   //  8192*256
  float* out = (float*)d_out;                // z_q (N*D) ++ loss (1)
  (void)in_sizes; (void)n_in; (void)out_size; (void)ws_size;

  // Workspace layout (all 256B-aligned):
  char* ws = (char*)d_ws;
  size_t off = 0;
  unsigned short* cb_hi = (unsigned short*)(ws + off); off += (size_t)VQ_K * VQ_DIM * 2;  //  4 MB
  unsigned short* cb_lo = (unsigned short*)(ws + off); off += (size_t)VQ_K * VQ_DIM * 2;  //  4 MB
  unsigned short* z_hi  = (unsigned short*)(ws + off); off += (size_t)VQ_N * VQ_DIM * 2;  // 16 MB
  unsigned short* z_lo  = (unsigned short*)(ws + off); off += (size_t)VQ_N * VQ_DIM * 2;  // 16 MB
  float* e_sq           = (float*)(ws + off);          off += (size_t)VQ_K * 4;           // 32 KB
  int*   idx            = (int*)(ws + off);            off += (size_t)VQ_N * 4;           // 128 KB
  float* partials       = (float*)(ws + off);          off += 1024 * 4;                   // 4 KB

  // 1) bf16 hi/lo splits.
  vq_split_kernel<<<2048, 256, 0, stream>>>(cb, cb_hi, cb_lo, VQ_K * VQ_DIM);
  vq_split_kernel<<<4096, 256, 0, stream>>>(z, z_hi, z_lo, VQ_N * VQ_DIM);
  // 2) ||e||^2.
  vq_esq_kernel<<<VQ_K, 256, 0, stream>>>(cb, e_sq);
  // 3) WMMA distance + argmin: 256 blocks x 8 waves x 16 rows = 32768 rows.
  vq_argmin_kernel<<<VQ_N / 128, 256, 0, stream>>>(z_hi, z_lo, cb_hi, cb_lo,
                                                   e_sq, idx);
  // 4) gather z_q + partial loss; 1024 blocks x 8192 elements.
  vq_gather_loss_kernel<<<1024, 256, 0, stream>>>(z, cb, idx, out, partials);
  // 5) deterministic scalar loss.
  vq_finalize_kernel<<<1, 1024, 0, stream>>>(partials, out + (size_t)VQ_N * VQ_DIM);
}
